// ExpertFFN_41343355191804
// MI455X (gfx1250) — matrix operand order repaired and checked by reading, not compiled, not run on hardware
//
#include <hip/hip_runtime.h>
#include <hip/hip_bf16.h>
#include <stdint.h>

// ---------------- problem constants (match reference) ----------------
static constexpr int H = 1024;   // hidden dim
static constexpr int E = 8;      // experts
static constexpr int TILE_M = 64;          // tokens per block tile (4 WMMA M-tiles)
static constexpr int NSPLIT = 4;           // N-sweep split for grid parallelism
static constexpr int FRAG_STRIDE = 1056;   // bytes per A fragment in LDS (32 lanes * 32B + pad)

typedef __attribute__((ext_vector_type(16))) __bf16 v16bf;
typedef __attribute__((ext_vector_type(8)))  float  v8f;
typedef __attribute__((ext_vector_type(4)))  float  f32x4;
typedef __attribute__((ext_vector_type(2)))  unsigned int u32x2;

union FragBF { f32x4 q[2]; v16bf v; };

__device__ __forceinline__ unsigned short f32_to_bf16_rne(float f) {
    union { float f; unsigned int u; } c; c.f = f;
    unsigned int u = c.u;
    u += 0x7FFFu + ((u >> 16) & 1u);   // round-to-nearest-even
    return (unsigned short)(u >> 16);
}

// ---------------------------------------------------------------------
// Kernel 1: transpose + convert expert weights
//   w  : (E, K=H, N=H) fp32 -> wt : (E, N=H, K=H) bf16
// ---------------------------------------------------------------------
__global__ void moe_transpose_w(const float* __restrict__ w,
                                unsigned short* __restrict__ wt) {
    __shared__ float tile[32][33];
    const int nTiles1D = H / 32;
    int bid = blockIdx.x;
    int e  = bid / (nTiles1D * nTiles1D);
    int r  = bid % (nTiles1D * nTiles1D);
    int kb = (r / nTiles1D) * 32;
    int nb = (r % nTiles1D) * 32;
    int tx = threadIdx.x & 31;
    int ty = threadIdx.x >> 5;
    const float* we = w + (size_t)e * H * H;
    unsigned short* wte = wt + (size_t)e * H * H;
    #pragma unroll
    for (int i = 0; i < 4; ++i)
        tile[ty + 8 * i][tx] = we[(size_t)(kb + ty + 8 * i) * H + (nb + tx)];
    __syncthreads();
    #pragma unroll
    for (int i = 0; i < 4; ++i) {
        int n = nb + ty + 8 * i;
        int k = kb + tx;
        wte[(size_t)n * H + k] = f32_to_bf16_rne(tile[tx][ty + 8 * i]);
    }
}

// ---------------------------------------------------------------------
// Kernel 2: router (one wave32 per token) + expert histogram
// ---------------------------------------------------------------------
__global__ void moe_router(const float* __restrict__ x,
                           const float* __restrict__ rw,   // (H, E)
                           const float* __restrict__ rb,   // (E)
                           float* __restrict__ gate,
                           int* __restrict__ idx,
                           int* __restrict__ counts,
                           int T) {
    int wave = threadIdx.x >> 5;
    int lane = threadIdx.x & 31;
    int t = blockIdx.x * 8 + wave;
    if (t >= T) return;

    float acc[E];
    #pragma unroll
    for (int e = 0; e < E; ++e) acc[e] = 0.0f;

    const float* xr = x + (size_t)t * H;
    for (int h = lane; h < H; h += 32) {
        float xv = xr[h];
        const float* wr = rw + (size_t)h * E;
        f32x4 w0 = *(const f32x4*)(wr);
        f32x4 w1 = *(const f32x4*)(wr + 4);
        acc[0] += xv * w0.x; acc[1] += xv * w0.y;
        acc[2] += xv * w0.z; acc[3] += xv * w0.w;
        acc[4] += xv * w1.x; acc[5] += xv * w1.y;
        acc[6] += xv * w1.z; acc[7] += xv * w1.w;
    }
    #pragma unroll
    for (int off = 16; off >= 1; off >>= 1) {
        #pragma unroll
        for (int e = 0; e < E; ++e)
            acc[e] += __shfl_xor(acc[e], off, 32);
    }
    if (lane == 0) {
        float lmax = -3.0e38f; int best = 0;
        #pragma unroll
        for (int e = 0; e < E; ++e) {
            float l = acc[e] + rb[e];
            acc[e] = l;
            if (l > lmax) { lmax = l; best = e; }
        }
        float s = 0.0f;
        #pragma unroll
        for (int e = 0; e < E; ++e) s += __expf(acc[e] - lmax);
        gate[t] = 1.0f / s;
        idx[t]  = best;
        atomicAdd(&counts[best], 1);
    }
}

// ---------------------------------------------------------------------
// Kernel 3a/3b/3c: init, scan, scatter
// ---------------------------------------------------------------------
__global__ void moe_init(int* counts, int* cursors, int* totalTiles) {
    int i = threadIdx.x;
    if (i < E) { counts[i] = 0; cursors[i] = 0; }
    if (i == 0) *totalTiles = 0;
}

__global__ void moe_scan(const int* __restrict__ counts,
                         int* __restrict__ offsets,
                         int* __restrict__ tilePrefix,
                         int* __restrict__ totalTiles) {
    if (threadIdx.x != 0 || blockIdx.x != 0) return;
    int off = 0, tp = 0;
    #pragma unroll
    for (int e = 0; e < E; ++e) {
        offsets[e] = off;        off += counts[e];
        tilePrefix[e] = tp;      tp  += (counts[e] + TILE_M - 1) / TILE_M;
    }
    offsets[E] = off;
    tilePrefix[E] = tp;
    *totalTiles = tp;
}

__global__ void moe_scatter(const int* __restrict__ idx,
                            const int* __restrict__ offsets,
                            int* __restrict__ cursors,
                            int* __restrict__ tokenList,
                            int T) {
    int t = blockIdx.x * blockDim.x + threadIdx.x;
    if (t >= T) return;
    int e = idx[t];
    int pos = atomicAdd(&cursors[e], 1);
    tokenList[offsets[e] + pos] = t;
}

// ---------------------------------------------------------------------
// Kernel 4: grouped expert GEMM (bf16 WMMA, fp32 acc)
//   block = one 64-token tile x one quarter of the N sweep (256 cols)
//   wave owns 16 columns, all 4 M-tiles -> one B load feeds 4 WMMAs.
//   B fragment double-buffered across K steps; A fragments loaded into
//   4 distinct register sets for pipelined DScnt waits.
//   Branch-free epilogue: pad rows store to a dump row.
// ---------------------------------------------------------------------
__global__ void moe_gemm(const float* __restrict__ x,
                         const unsigned short* __restrict__ wt,  // (E,N,K) bf16, padded
                         const float* __restrict__ eb,           // (E,H)
                         const float* __restrict__ gate,
                         const int* __restrict__ tokenList,
                         const int* __restrict__ counts,
                         const int* __restrict__ offsets,
                         const int* __restrict__ tilePrefix,
                         const int* __restrict__ totalTiles,
                         float* __restrict__ out,
                         float* __restrict__ dumpRow) {
    extern __shared__ unsigned char smem[];
    unsigned char* Afrag = smem;                               // 128 frags * FRAG_STRIDE
    u32x2* sPair = (u32x2*)(smem + 128 * FRAG_STRIDE);         // 64 x {token, gate}

    int tile = blockIdx.x >> 2;          // 64-token tile id
    int ns   = blockIdx.x & 3;           // which quarter of N
    if (tile >= *totalTiles) return;

    int e = 0;
    #pragma unroll
    for (int i = 0; i < E - 1; ++i)
        if (tile >= tilePrefix[i + 1]) e = i + 1;
    int lt    = tile - tilePrefix[e];
    int base  = offsets[e] + lt * TILE_M;
    int valid = counts[e] - lt * TILE_M;
    if (valid > TILE_M) valid = TILE_M;

    int wave = threadIdx.x >> 5;
    int lane = threadIdx.x & 31;

    // ---- stage {token, gate} pairs into LDS
    if (threadIdx.x < TILE_M) {
        int r = threadIdx.x;
        int tok = (r < valid) ? tokenList[base + r] : -1;
        float g = (tok >= 0) ? gate[tok] : 0.0f;
        u32x2 pr;
        pr.x = (unsigned int)tok;
        pr.y = __float_as_uint(g);
        sPair[r] = pr;
    }

    // ---- stage A (64 rows x 1024 K) into LDS, bf16, fragment-linear layout.
    #pragma unroll 1
    for (int rr = 0; rr < 8; ++rr) {
        int m = wave * 8 + rr;
        int tok = (m < valid) ? tokenList[base + m] : -1;
        const float* xr = x + (size_t)(tok < 0 ? 0 : tok) * H;
        int mtile = m >> 4, mrow = m & 15;
        #pragma unroll
        for (int j = 0; j < 8; ++j) {
            int k0 = 4 * lane + 128 * j;
            f32x4 v = {0.f, 0.f, 0.f, 0.f};
            if (tok >= 0) v = *(const f32x4*)(xr + k0);
            // ISA 7.12.2 A layout: lane-in-frag = mrow + 16*((k>>3)&1),
            // halfword pos = (k&7) + 8*((k>>4)&1); 4 consecutive k share a lane.
            int kk  = k0 & 31;
            int lif = mrow + 16 * ((kk >> 3) & 1);
            int pos = (kk & 7) + 8 * ((kk >> 4) & 1);
            int fragid = (k0 >> 5) * 4 + mtile;
            unsigned short h0 = f32_to_bf16_rne(v.x);
            unsigned short h1 = f32_to_bf16_rne(v.y);
            unsigned short h2 = f32_to_bf16_rne(v.z);
            unsigned short h3 = f32_to_bf16_rne(v.w);
            u32x2 packed;
            packed.x = (unsigned int)h0 | ((unsigned int)h1 << 16);
            packed.y = (unsigned int)h2 | ((unsigned int)h3 << 16);
            *(u32x2*)(Afrag + (size_t)fragid * FRAG_STRIDE + lif * 32 + 2 * pos) = packed;
        }
    }
    __syncthreads();

    int half = lane >> 4;                 // C/B K-half & M-half selector
    int l15  = lane & 15;
    const unsigned char* afBase = Afrag + lane * 32;

    #pragma unroll 1
    for (int cg = 0; cg < (H / NSPLIT) / 128; ++cg) {        // 2 passes
        int n = ns * (H / NSPLIT) + cg * 128 + wave * 16 + l15;
        const unsigned short* bp = wt + ((size_t)e * H + n) * H + half * 8;

        v8f acc[4];
        #pragma unroll
        for (int mt = 0; mt < 4; ++mt)
            acc[mt] = (v8f){0.f, 0.f, 0.f, 0.f, 0.f, 0.f, 0.f, 0.f};

        // prime B double buffer
        FragBF b;
        b.q[0] = *(const f32x4*)(bp);
        b.q[1] = *(const f32x4*)(bp + 16);

        #pragma unroll 2
        for (int kb = 0; kb < H; kb += 32) {
            // prefetch next B fragment (wt padded; last prefetch is dead data)
            FragBF bnext;
            bnext.q[0] = *(const f32x4*)(bp + kb + 32);
            bnext.q[1] = *(const f32x4*)(bp + kb + 48);

            const unsigned char* af = afBase + (size_t)(kb >> 5) * 4 * FRAG_STRIDE;
            FragBF a0, a1, a2, a3;
            a0.q[0] = *(const f32x4*)(af);
            a0.q[1] = *(const f32x4*)(af + 16);
            a1.q[0] = *(const f32x4*)(af + FRAG_STRIDE);
            a1.q[1] = *(const f32x4*)(af + FRAG_STRIDE + 16);
            a2.q[0] = *(const f32x4*)(af + 2 * FRAG_STRIDE);
            a2.q[1] = *(const f32x4*)(af + 2 * FRAG_STRIDE + 16);
            a3.q[0] = *(const f32x4*)(af + 3 * FRAG_STRIDE);
            a3.q[1] = *(const f32x4*)(af + 3 * FRAG_STRIDE + 16);

            acc[0] = __builtin_amdgcn_wmma_f32_16x16x32_bf16(
                         false, a0.v, false, b.v, (short)0, acc[0], false, false);
            acc[1] = __builtin_amdgcn_wmma_f32_16x16x32_bf16(
                         false, a1.v, false, b.v, (short)0, acc[1], false, false);
            acc[2] = __builtin_amdgcn_wmma_f32_16x16x32_bf16(
                         false, a2.v, false, b.v, (short)0, acc[2], false, false);
            acc[3] = __builtin_amdgcn_wmma_f32_16x16x32_bf16(
                         false, a3.v, false, b.v, (short)0, acc[3], false, false);
            b = bnext;
        }

        // ---- branch-free epilogue: pad rows redirect to dump row
        float bn = eb[(size_t)e * H + n];
        #pragma unroll
        for (int mt = 0; mt < 4; ++mt) {
            #pragma unroll
            for (int r = 0; r < 8; ++r) {
                int m = mt * 16 + half * 8 + r;      // C: M = (lane>>4)*8 + vgpr
                u32x2 pr = sPair[m];
                int tok = (int)pr.x;
                float g = __uint_as_float(pr.y);
                float* obase = (tok >= 0) ? (out + (size_t)tok * H) : dumpRow;
                obase[n] = g * (acc[mt][r] + bn);
            }
        }
    }
}

// ---------------------------------------------------------------------
extern "C" void kernel_launch(void* const* d_in, const int* in_sizes, int n_in,
                              void* d_out, int out_size, void* d_ws, size_t ws_size,
                              hipStream_t stream) {
    const float* x  = (const float*)d_in[0];   // (B,S,H)
    const float* rw = (const float*)d_in[1];   // (H,E)
    const float* rb = (const float*)d_in[2];   // (E)
    const float* ew = (const float*)d_in[3];   // (E,H,H)
    const float* eb = (const float*)d_in[4];   // (E,H)
    float* out = (float*)d_out;

    const int T = in_sizes[0] / H;             // 8192 tokens

    // ---- carve workspace ----
    uint8_t* p = (uint8_t*)d_ws;
    unsigned short* wt = (unsigned short*)p;
    p += ((size_t)E * H * H + 4096) * sizeof(unsigned short);  // 16 MB + prefetch pad
    float* gate        = (float*)p;           p += (size_t)T * sizeof(float);
    int*   idx         = (int*)p;             p += (size_t)T * sizeof(int);
    int*   tokenList   = (int*)p;             p += (size_t)T * sizeof(int);
    int*   counts      = (int*)p;             p += E * sizeof(int);
    int*   cursors     = (int*)p;             p += E * sizeof(int);
    int*   offsets     = (int*)p;             p += (E + 1) * sizeof(int);
    int*   tilePrefix  = (int*)p;             p += (E + 1) * sizeof(int);
    int*   totalTiles  = (int*)p;             p += 16 * sizeof(int);
    float* dumpRow     = (float*)p;           p += (size_t)H * sizeof(float);

    // 1) weight transpose/convert
    moe_transpose_w<<<E * (H / 32) * (H / 32), 256, 0, stream>>>(ew, wt);

    // 2) init counters, then router
    moe_init<<<1, 32, 0, stream>>>(counts, cursors, totalTiles);
    moe_router<<<(T + 7) / 8, 256, 0, stream>>>(x, rw, rb, gate, idx, counts, T);

    // 3) scan + scatter
    moe_scan<<<1, 32, 0, stream>>>(counts, offsets, tilePrefix, totalTiles);
    moe_scatter<<<(T + 255) / 256, 256, 0, stream>>>(idx, offsets, cursors, tokenList, T);

    // 4) grouped GEMM: worst case 64-token tile count, x4 N-split
    int maxTiles = T / TILE_M + E;                        // 136
    size_t shmem = 128 * FRAG_STRIDE + TILE_M * sizeof(u32x2);
    moe_gemm<<<maxTiles * NSPLIT, 256, shmem, stream>>>(x, wt, eb, gate, tokenList,
                                                        counts, offsets, tilePrefix,
                                                        totalTiles, out, dumpRow);
}
